// FakeGptOssExperts_45853070852311
// MI455X (gfx1250) — matrix operand order repaired and checked by reading, not compiled, not run on hardware
//
#include <hip/hip_runtime.h>
#include <hip/hip_bf16.h>

#define ALPHA  1.702f
#define LIMIT  7.0f
#define NE     16
#define NT     1024
#define NH     2880
#define NI     2880
#define TWO_I  5760

typedef __attribute__((ext_vector_type(16))) __bf16 v16bf;
typedef __attribute__((ext_vector_type(8)))  __bf16 v8bf;
typedef __attribute__((ext_vector_type(8)))  float  v8f;
typedef __attribute__((ext_vector_type(4)))  unsigned int v4u;
typedef __attribute__((ext_vector_type(8)))  int v8i;
typedef __attribute__((ext_vector_type(4)))  int v4i;

#define LDS_STRIDE 40   // elements per LDS row: 32 K + 8 pad = 80 bytes (16B aligned)

// TDM group1 dword0: workgroup_mask=0, data_size=1 (2 bytes) [17:16],
// pad_enable [20], pad_interval=3 (16 DWORDs = 64B = one 32-elt bf16 row) [24:22],
// pad_amount=3 (4 DWORDs = 16B) [31:25]  ->  LDS row stride 80B == LDS_STRIDE
#define TDM_G1W0_PAD80 0x06D10000u

// ---- CDNA5 async copy: global -> LDS, 16B per lane, tracked by ASYNCcnt ----
__device__ __forceinline__ void async_copy_b128(unsigned lds_off, const void* gptr) {
  unsigned long long ga = (unsigned long long)(size_t)gptr;
  asm volatile("global_load_async_to_lds_b128 %0, %1, off"
               :: "v"(lds_off), "v"(ga)
               : "memory");
}
__device__ __forceinline__ void wait_async0() {
  asm volatile("s_wait_asynccnt 0x0" ::: "memory");
}
// LDS offset (addr[31:0] of the flat address of a __shared__ object)
__device__ __forceinline__ unsigned lds_off_of(const void* p) {
  return (unsigned)(size_t)p;
}

// ---- CDNA5 Tensor Data Mover: 2D tile (tile_d0 x tile_d1) global -> LDS ----
// Descriptor per CDNA5 ISA (groups 2/3 zero: <=2D tensor).
// 6-arg builtin form (clang-23 / therock-10.0 headers).
__device__ __forceinline__ void tdm_load_2d(unsigned lds_addr, const void* gaddr,
                                            unsigned tensor_d0, unsigned tensor_d1,
                                            unsigned stride_d0_elts,
                                            unsigned tile_d0, unsigned tile_d1) {
  unsigned long long ga = (unsigned long long)(size_t)gaddr;
  v4u g0;
  g0.x = 0x1u;                                             // count=1, user mode
  g0.y = lds_addr;                                         // LDS byte address
  g0.z = (unsigned)ga;                                     // global addr [31:0]
  g0.w = ((unsigned)(ga >> 32) & 0x1FFFFFFu) | (2u << 30); // addr[56:32] | type=2
  v8i g1;
  g1[0] = (int)TDM_G1W0_PAD80;                                           // flags
  g1[1] = (int)((tensor_d0 & 0xffffu) << 16);                            // dim0 lo
  g1[2] = (int)(((tensor_d0 >> 16) & 0xffffu) | ((tensor_d1 & 0xffffu) << 16));
  g1[3] = (int)(((tensor_d1 >> 16) & 0xffffu) | ((tile_d0 & 0xffffu) << 16));
  g1[4] = (int)(tile_d1 & 0xffffu);                                      // tile_d1
  g1[5] = (int)stride_d0_elts;                                           // stride lo
  g1[6] = 0;
  g1[7] = 0;
  v4i z4 = (v4i)0;
  v8i z8 = (v8i)0;
  __builtin_amdgcn_tensor_load_to_lds(g0, g1, z4, z4, z8, 0);
}
__device__ __forceinline__ void wait_tensor0() {
  __builtin_amdgcn_s_wait_tensorcnt(0);
}

// A-fragment (16x32 bf16, row-major in LDS): per ISA 7.12.2
// lanes 0-15: M=lane, K=0..7 & 16..23 ; lanes 16-31: M=lane-16, K=8..15 & 24..31
__device__ __forceinline__ v16bf frag_a(const unsigned short* base, int lane) {
  const int m  = lane & 15;
  const int ko = (lane < 16) ? 0 : 8;
  const __bf16* p = (const __bf16*)(base + m * LDS_STRIDE + ko);
  v8bf lo = *(const v8bf*)(p);        // 16B aligned
  v8bf hi = *(const v8bf*)(p + 16);   // 16B aligned
  return __builtin_shufflevector(lo, hi, 0,1,2,3,4,5,6,7,8,9,10,11,12,13,14,15);
}

// B-fragment (32x16 bf16), LDS holds the tile N-major (ldsB[n][k]):
// lanes 0-15: N=lane, K=0..7 & 16..23 ; lanes 16-31: N=lane-16, K=8..15 & 24..31
__device__ __forceinline__ v16bf frag_b(const unsigned short* base, int lane) {
  const int n  = lane & 15;
  const int ko = (lane < 16) ? 0 : 8;
  const __bf16* p = (const __bf16*)(base + n * LDS_STRIDE + ko);
  v8bf lo = *(const v8bf*)(p);
  v8bf hi = *(const v8bf*)(p + 16);
  return __builtin_shufflevector(lo, hi, 0,1,2,3,4,5,6,7,8,9,10,11,12,13,14,15);
}

__device__ __forceinline__ v8f wmma_bf16(v16bf a, v16bf b, v8f c) {
  return __builtin_amdgcn_wmma_f32_16x16x32_bf16(false, a, false, b, (short)0, c, false, false);
}

// ---------------------------------------------------------------------------
// GEMM1 + bias + gated activation -> h (E,T,I) bf16 in workspace
// block: 128 tokens x 64 h-cols (== 128 interleaved gate_up cols)
// A tile staged by the Tensor Data Mover (wave 0 issues, TENSORcnt tracked).
// ---------------------------------------------------------------------------
__global__ __launch_bounds__(256) void gptoss_gemm1_gate(
    const unsigned short* __restrict__ x,     // (E,T,H) bf16 bits
    const unsigned short* __restrict__ w,     // (E,H,2I) bf16 bits
    const unsigned short* __restrict__ bias,  // (E,2I) bf16 bits
    __bf16* __restrict__ hbuf)                // (E,T,I)
{
  __shared__ unsigned short ldsA[128 * LDS_STRIDE]; // 128 rows x 32 k (TDM-padded)
  __shared__ unsigned short ldsG[64 * LDS_STRIDE];  // 64 i-cols x 32 k (gate)
  __shared__ unsigned short ldsU[64 * LDS_STRIDE];  // 64 i-cols x 32 k (up)

  const int e    = blockIdx.z;
  const int mb   = blockIdx.y;   // 0..7  token blocks of 128
  const int nb   = blockIdx.x;   // 0..44 i blocks of 64
  const int tid  = threadIdx.x;
  const int lane = tid & 31;
  const int wave = tid >> 5;     // 8 waves
  const int wm   = wave & 3;     // wave row: 32 tokens each
  const int wn   = wave >> 2;    // wave col: 32 i each

  const unsigned short* X = x + ((size_t)e * NT + (size_t)mb * 128) * NH;
  const unsigned short* W = w + (size_t)e * NH * TWO_I + (size_t)nb * 128;
  const unsigned ldsA_base = lds_off_of(ldsA);

  v8f accG[2][2], accU[2][2];
#pragma unroll
  for (int m = 0; m < 2; ++m)
#pragma unroll
    for (int n = 0; n < 2; ++n) { accG[m][n] = (v8f)0.0f; accU[m][n] = (v8f)0.0f; }

  for (int k0 = 0; k0 < NH; k0 += 32) {
    // ---- stage A: one TDM op moves the whole 128x32 tile (padded rows)
    if (wave == 0) {
      tdm_load_2d(ldsA_base, X + k0,
                  /*tensor_d0=*/NH, /*tensor_d1=*/128,
                  /*stride_d0=*/NH,
                  /*tile_d0=*/32, /*tile_d1=*/128);
    }
    // ---- stage B: 32x128 gate_up cols, transpose + gate/up deinterleave
#pragma unroll
    for (int it = 0; it < 2; ++it) {
      const int c  = tid + it * 256;
      const int k  = c >> 4;
      const int cg = c & 15;
      uint4 v = *(const uint4*)(W + (size_t)(k0 + k) * TWO_I + cg * 8);
      const unsigned short* s = (const unsigned short*)&v;
#pragma unroll
      for (int j = 0; j < 8; ++j) {
        const int col = cg * 8 + j;
        const int i   = col >> 1;
        if (col & 1) ldsU[i * LDS_STRIDE + k] = s[j];
        else         ldsG[i * LDS_STRIDE + k] = s[j];
      }
    }
    if (wave == 0) wait_tensor0();
    __syncthreads();

    v16bf aF[2], gF[2], uF[2];
#pragma unroll
    for (int m = 0; m < 2; ++m)
      aF[m] = frag_a(ldsA + (wm * 32 + m * 16) * LDS_STRIDE, lane);
#pragma unroll
    for (int n = 0; n < 2; ++n) {
      gF[n] = frag_b(ldsG + (wn * 32 + n * 16) * LDS_STRIDE, lane);
      uF[n] = frag_b(ldsU + (wn * 32 + n * 16) * LDS_STRIDE, lane);
    }
#pragma unroll
    for (int m = 0; m < 2; ++m)
#pragma unroll
      for (int n = 0; n < 2; ++n) {
        accG[m][n] = wmma_bf16(aF[m], gF[n], accG[m][n]);
        accU[m][n] = wmma_bf16(aF[m], uF[n], accU[m][n]);
      }
    __syncthreads();
  }

  // ---- epilogue: bias + clamp + glu, write bf16 h
#pragma unroll
  for (int n = 0; n < 2; ++n) {
    const int ig = nb * 64 + wn * 32 + n * 16 + (lane & 15);
    // gate bias at col 2*ig, up bias at 2*ig+1 : one aligned b32 load
    const unsigned int bb = *(const unsigned int*)(bias + (size_t)e * TWO_I + 2 * ig);
    const float gb = __uint_as_float((bb & 0xffffu) << 16);
    const float ub = __uint_as_float(bb & 0xffff0000u);
#pragma unroll
    for (int m = 0; m < 2; ++m) {
      const int rbase = mb * 128 + wm * 32 + m * 16 + ((lane < 16) ? 0 : 8);
#pragma unroll
      for (int r = 0; r < 8; ++r) {
        float g = accG[m][n][r] + gb;
        float u = accU[m][n][r] + ub;
        g = fminf(g, LIMIT);
        u = fminf(fmaxf(u, -LIMIT), LIMIT);
        const float glu = g / (1.0f + __expf(-ALPHA * g));
        const float hv  = (u + 1.0f) * glu;
        hbuf[((size_t)e * NT + (rbase + r)) * NI + ig] = (__bf16)hv;
      }
    }
  }
}

// ---------------------------------------------------------------------------
// GEMM2 + bias -> out (E,T,H) bf16
// block: 128 tokens x 64 out-cols; A tile staged with async-to-LDS (ASYNCcnt).
// ---------------------------------------------------------------------------
__global__ __launch_bounds__(256) void gptoss_gemm2_down(
    const unsigned short* __restrict__ h,     // (E,T,I) bf16 bits (workspace)
    const unsigned short* __restrict__ w,     // (E,I,H) bf16 bits
    const unsigned short* __restrict__ bias,  // (E,H) bf16 bits
    __bf16* __restrict__ out)                 // (E,T,H)
{
  __shared__ unsigned short ldsA[128 * LDS_STRIDE]; // 128 rows x 32 k
  __shared__ unsigned short ldsB[64 * LDS_STRIDE];  // 64 cols x 32 k (transposed)

  const int e    = blockIdx.z;
  const int mb   = blockIdx.y;   // 0..7  token blocks of 128
  const int nb   = blockIdx.x;   // 0..44 H blocks of 64
  const int tid  = threadIdx.x;
  const int lane = tid & 31;
  const int wave = tid >> 5;
  const int wm   = wave & 3;
  const int wn   = wave >> 2;

  const unsigned short* A = h + ((size_t)e * NT + (size_t)mb * 128) * NI;
  const unsigned short* W = w + (size_t)e * NI * NH + (size_t)nb * 64;

  v8f acc[2][2];
#pragma unroll
  for (int m = 0; m < 2; ++m)
#pragma unroll
    for (int n = 0; n < 2; ++n) acc[m][n] = (v8f)0.0f;

  for (int k0 = 0; k0 < NI; k0 += 32) {
    // ---- stage A: 128x32, 2 chunks per thread, async -> LDS
#pragma unroll
    for (int it = 0; it < 2; ++it) {
      const int c   = tid + it * 256;
      const int row = c >> 2;
      const int kg  = c & 3;
      async_copy_b128(lds_off_of(ldsA + row * LDS_STRIDE + kg * 8),
                      A + (size_t)row * NI + (k0 + kg * 8));
    }
    // ---- stage B: 32 k x 64 cols, transpose-on-store, 1 chunk per thread
    {
      const int c  = tid;
      const int k  = c >> 3;
      const int cg = c & 7;
      uint4 v = *(const uint4*)(W + (size_t)(k0 + k) * NH + cg * 8);
      const unsigned short* s = (const unsigned short*)&v;
#pragma unroll
      for (int j = 0; j < 8; ++j)
        ldsB[(cg * 8 + j) * LDS_STRIDE + k] = s[j];
    }
    wait_async0();
    __syncthreads();

    v16bf aF[2], bF[2];
#pragma unroll
    for (int m = 0; m < 2; ++m)
      aF[m] = frag_a(ldsA + (wm * 32 + m * 16) * LDS_STRIDE, lane);
#pragma unroll
    for (int n = 0; n < 2; ++n)
      bF[n] = frag_b(ldsB + (wn * 32 + n * 16) * LDS_STRIDE, lane);
#pragma unroll
    for (int m = 0; m < 2; ++m)
#pragma unroll
      for (int n = 0; n < 2; ++n)
        acc[m][n] = wmma_bf16(aF[m], bF[n], acc[m][n]);
    __syncthreads();
  }

  // ---- epilogue: + bias, store bf16
#pragma unroll
  for (int n = 0; n < 2; ++n) {
    const int col = nb * 64 + wn * 32 + n * 16 + (lane & 15);
    const unsigned int bbits = bias[(size_t)e * NH + col];
    const float bv = __uint_as_float(bbits << 16);
#pragma unroll
    for (int m = 0; m < 2; ++m) {
      const int rbase = mb * 128 + wm * 32 + m * 16 + ((lane < 16) ? 0 : 8);
#pragma unroll
      for (int r = 0; r < 8; ++r) {
        const float ov = acc[m][n][r] + bv;
        out[((size_t)e * NT + (rbase + r)) * NH + col] = (__bf16)ov;
      }
    }
  }
}

extern "C" void kernel_launch(void* const* d_in, const int* in_sizes, int n_in,
                              void* d_out, int out_size, void* d_ws, size_t ws_size,
                              hipStream_t stream) {
  (void)in_sizes; (void)n_in; (void)out_size; (void)ws_size;
  const unsigned short* x    = (const unsigned short*)d_in[0]; // (E,T,H)   bf16
  const unsigned short* gup  = (const unsigned short*)d_in[1]; // (E,H,2I)  bf16
  const unsigned short* gupb = (const unsigned short*)d_in[2]; // (E,2I)    bf16
  const unsigned short* dp   = (const unsigned short*)d_in[3]; // (E,I,H)   bf16
  const unsigned short* dpb  = (const unsigned short*)d_in[4]; // (E,H)     bf16

  __bf16* hbuf = (__bf16*)d_ws;            // (E,T,I) bf16 = 94.4 MB scratch
  __bf16* out  = (__bf16*)d_out;           // (E,T,H) bf16

  dim3 block(256, 1, 1);
  dim3 grid1(NI / 64, NT / 128, NE);       // 45 x 8 x 16
  gptoss_gemm1_gate<<<grid1, block, 0, stream>>>(x, gup, gupb, hbuf);

  dim3 grid2(NH / 64, NT / 128, NE);       // 45 x 8 x 16
  gptoss_gemm2_down<<<grid2, block, 0, stream>>>((const unsigned short*)hbuf, dp, dpb, out);
}
